// TheRecurrentNet_90056874262605
// MI455X (gfx1250) — matrix operand order repaired and checked by reading, not compiled, not run on hardware
//
#include <hip/hip_runtime.h>
#include <cstdint>
#include <cstddef>

typedef __bf16 bf16_t;
typedef __attribute__((ext_vector_type(16))) __bf16 v16bf;
typedef __attribute__((ext_vector_type(8)))  float  v8f;
typedef unsigned short u16;

#define HH 500   // true hidden width
#define HP 512   // padded hidden width

__device__ __forceinline__ unsigned short f2bf(float f) {
  unsigned int u = __float_as_uint(f);
  u += 0x7fffu + ((u >> 16) & 1u);
  return (unsigned short)(u >> 16);
}
__device__ __forceinline__ float sigmoidf_(float x) { return 1.f / (1.f + __expf(-x)); }
__device__ __forceinline__ v8f asf(v16bf x) { return __builtin_bit_cast(v8f, x); }

// ---------------------------------------------------------------------------
// A-matrix fragment swizzle (16-bit WMMA A layout, K split 8+8 per lane-half).
// Logical (row r, col c) of a row-major [rows x 32*nkt] matrix -> flat index in
// fragment-major storage: tile (r/16, c/32), lane = (r&15) + kh*16, 16 elems
// per lane contiguous.  One wave's A fragment = 1KB contiguous.
// ---------------------------------------------------------------------------
__device__ __forceinline__ long long aswz(long long r, int c, int nkt) {
  int kt = c >> 5, c32 = c & 31;
  int kh = (c32 >> 3) & 1;                    // K-half
  int e  = (c32 & 7) | ((c32 >> 4) << 3);     // element within lane (+8 if c32>=16)
  int lane = (int)(r & 15) + kh * 16;
  return (((r >> 4) * (long long)nkt + kt) * 32 + lane) * 16 + e;
}

// ---------------------------------------------------------------------------
// fp32 -> bf16 activation convert into A-fragment-major layout (zero padded)
// ---------------------------------------------------------------------------
__global__ void conv_act_kernel(const float* __restrict__ src, int src_ld, int valid,
                                u16* __restrict__ dst, int dst_ld, long long rows) {
  long long idx = (long long)blockIdx.x * blockDim.x + threadIdx.x;
  long long total = rows * dst_ld;
  if (idx >= total) return;
  int c = (int)(idx % dst_ld);
  long long r = idx / dst_ld;
  float v = (c < valid) ? src[r * src_ld + c] : 0.f;
  dst[aswz(r, c, dst_ld >> 5)] = f2bf(v);
}

// ---------------------------------------------------------------------------
// fp32 row-major weight (K x N) -> bf16 B-fragment-major (K split 8+8 per
// lane-half): fragment (ntg, kt): lane l holds col n = ntg*16+(l&15),
// k = kt*32 + (l>>4)*8 + (e&7) + (e>>3)*16, 16 elems contiguous.  kt_off supports concat-K.
// ---------------------------------------------------------------------------
__global__ void conv_wfrag_kernel(const float* __restrict__ src, int src_ld, int kvalid, int nvalid,
                                  u16* __restrict__ dst, int nkt_total, int kt_off, int nkt_zone, int ntn) {
  long long idx = (long long)blockIdx.x * blockDim.x + threadIdx.x;
  long long total = (long long)ntn * nkt_zone * 512;
  if (idx >= total) return;
  int e = (int)(idx & 15);
  int l = (int)((idx >> 4) & 31);
  long long t = idx >> 9;
  int kt  = (int)(t % nkt_zone);
  int ntg = (int)(t / nkt_zone);
  int k = kt * 32 + (l >> 4) * 8 + (e & 7) + (e >> 3) * 16;
  int n = ntg * 16 + (l & 15);
  float v = (k < kvalid && n < nvalid) ? src[(long long)k * src_ld + n] : 0.f;
  dst[(((long long)ntg * nkt_total + kt_off + kt) * 32 + l) * 16 + e] = f2bf(v);
}

__global__ void zero_kernel(float* __restrict__ p, long long n) {
  long long i = (long long)blockIdx.x * blockDim.x + threadIdx.x;
  if (i < n) p[i] = 0.f;
}

// ---------------------------------------------------------------------------
// attention_a[r] = sigmoid( state[r,:500].w3[:500] + a[r/16,:500].w3[500:] + b3 )
// one wave per row
// ---------------------------------------------------------------------------
__global__ void att_kernel(const float* __restrict__ state, const float* __restrict__ a32,
                           const float* __restrict__ w3, const float* __restrict__ b3,
                           float* __restrict__ att, int M) {
  int wid = blockIdx.x * (blockDim.x >> 5) + (threadIdx.x >> 5);
  int lane = threadIdx.x & 31;
  if (wid >= M) return;
  const float* srow = state + (long long)wid * HH;
  const float* arow = a32 + (long long)(wid >> 4) * HP;
  float s = 0.f;
  for (int c = lane; c < HH; c += 32) s += srow[c] * w3[c] + arow[c] * w3[HH + c];
  for (int o = 1; o < 32; o <<= 1) s += __shfl_xor(s, o, 32);
  if (lane == 0) att[wid] = sigmoidf_(s + b3[0]);
}

// ---------------------------------------------------------------------------
// core[r] = bf16(relu(LN(P[bi] + Q[bjj] + fc2_b)))  with r = (b*16+i)*15 + j
// one wave per core row; output written in A-fragment-major layout
// ---------------------------------------------------------------------------
__global__ void core_kernel(const float* __restrict__ P, const float* __restrict__ Q,
                            const float* __restrict__ b2, const float* __restrict__ g2,
                            const float* __restrict__ bb2, u16* __restrict__ core_bf, int M) {
  int wid = blockIdx.x * (blockDim.x >> 5) + (threadIdx.x >> 5);
  int lane = threadIdx.x & 31;
  if (wid >= M) return;
  int b   = wid / 240;       // 16*15
  int rem = wid % 240;
  int i = rem / 15, j = rem % 15;
  int jj = j + (j >= i ? 1 : 0);
  const float* p = P + (long long)(b * 16 + i)  * HP;
  const float* q = Q + (long long)(b * 16 + jj) * HP;
  float vals[16];
  float s1 = 0.f, s2 = 0.f;
#pragma unroll
  for (int it = 0; it < 16; ++it) {
    int c = lane + it * 32;
    float x = 0.f;
    if (c < HH) { x = p[c] + q[c] + b2[c]; s1 += x; s2 += x * x; }
    vals[it] = x;
  }
  for (int o = 1; o < 32; o <<= 1) { s1 += __shfl_xor(s1, o, 32); s2 += __shfl_xor(s2, o, 32); }
  float mu = s1 / (float)HH;
  float var = s2 / (float)HH - mu * mu;
  float rstd = rsqrtf(var + 1e-5f);
#pragma unroll
  for (int it = 0; it < 16; ++it) {
    int c = lane + it * 32;
    float y = 0.f;
    if (c < HH) { float x = (vals[it] - mu) * rstd * g2[c] + bb2[c]; y = fmaxf(x, 0.f); }
    core_bf[aswz((long long)wid, c, HP >> 5)] = f2bf(y);
  }
}

// ---------------------------------------------------------------------------
// Generic bf16 WMMA GEMM + fused row epilogue.
//   C[m, n] = sum_k Aseg(m,k) * B[n, k]
//   A buffers are A-fragment-major; B buffers are B-fragment-major, so every
//   fragment load is one fully-coalesced 1KB contiguous wave access.
//   Aseg: kt <  nkt1 -> A1 (row m)
//         kt >= nkt1 -> A2 (row m / rowdiv2)
// Workgroup: 16 rows x full N.  N/128 waves; each wave owns 128 columns
// (8 WMMA n-tiles).  An empty asm barrier pins A + all 8 B fragments live
// between the load group and the WMMA group -> batched load clauses with
// staged s_wait_loadcnt.  __launch_bounds__(128, 4) raises the VGPR budget
// to 256/wave (4 waves/SIMD) so the ~230 live VGPRs don't spill to scratch.
// Epilogue: x = C + bias; optional LN(g,b) over ncols; act (0/relu/tanh);
// optional per-row scale; outputs: bf16 (A-fragment-major, zero-padded), fp32,
// atomic scatter-add (row m -> m/atom_div), or per-row dot+sigmoid.
// ---------------------------------------------------------------------------
__global__ void __launch_bounds__(128, 4) gemm_epi_kernel(
    const u16* __restrict__ A1_, int nkt1,
    const u16* __restrict__ A2_, int rowdiv2,
    const u16* __restrict__ B_, int Ktot,
    int N, int ncols,
    const float* __restrict__ bias,
    const float* __restrict__ ln_g, const float* __restrict__ ln_b,
    int act,
    const float* __restrict__ row_scale,
    u16* __restrict__ out_bf, long long ldob,
    float* __restrict__ out_f32, long long ldof,
    float* __restrict__ atom_out, int atom_div, long long ld_atom,
    const float* __restrict__ dotw, const float* __restrict__ dotb,
    float* __restrict__ dot_out) {
  extern __shared__ float tile[];
  const int ldt = N + 4;
  const int tid = threadIdx.x;
  const int lane = tid & 31;
  const int wave = tid >> 5;
  const int m0 = blockIdx.x * 16;
  const int row_in = lane & 15;
  const int kh = lane >> 4;            // K-half selector
  const int m = m0 + row_in;

  const bf16_t* A1 = (const bf16_t*)A1_;
  const bf16_t* A2 = (const bf16_t*)A2_;
  const bf16_t* B  = (const bf16_t*)B_;

  const int nkt  = Ktot >> 5;
  const int nkt2 = nkt - nkt1;

  // Fragment base pointers (lane-resolved); per K-tile step = +512 elems (1KB)
  const bf16_t* a1base = A1 + (((long long)(m >> 4) * nkt1) * 32 + lane) * 16;
  const bf16_t* a2base = nullptr;
  if (A2) {
    long long ra = m / rowdiv2;
    a2base = A2 + (((ra >> 4) * (long long)nkt2) * 32 + (int)(ra & 15) + kh * 16) * 16;
  }
  // B fragment base for this wave's first n-tile
  const bf16_t* bb = B + ((long long)(wave * 8) * nkt) * 512 + lane * 16;

  v8f acc[8];
#pragma unroll
  for (int t = 0; t < 8; ++t) acc[t] = (v8f){0.f, 0.f, 0.f, 0.f, 0.f, 0.f, 0.f, 0.f};

  for (int kt = 0; kt < nkt; ++kt) {
    // Prefetch A stream a few K-tiles ahead (global_prefetch_b8)
    {
      int kp = kt + 4;
      if (kp < nkt) {
        const bf16_t* pp = (kp < nkt1) ? a1base + (long long)kp * 512
                                       : a2base + (long long)(kp - nkt1) * 512;
        __builtin_prefetch((const void*)pp, 0, 1);
      }
    }
    v16bf af = (kt < nkt1) ? *(const v16bf*)(a1base + (long long)kt * 512)
                           : *(const v16bf*)(a2base + (long long)(kt - nkt1) * 512);
    // Batch-load all 8 B fragments into distinct registers first
    v16bf bfr[8];
#pragma unroll
    for (int nt = 0; nt < 8; ++nt)
      bfr[nt] = *(const v16bf*)(bb + ((long long)nt * nkt + kt) * 512);
    // Pin all fragments live here: loads cannot sink past this point and the
    // allocator must keep 9 distinct fragment register tuples -> one batched
    // clause of loads with staged waits instead of load->wait(0)->wmma x8.
    asm volatile("" :: "v"(asf(af)),
                 "v"(asf(bfr[0])), "v"(asf(bfr[1])), "v"(asf(bfr[2])), "v"(asf(bfr[3])),
                 "v"(asf(bfr[4])), "v"(asf(bfr[5])), "v"(asf(bfr[6])), "v"(asf(bfr[7])));
    // Then issue the 8 WMMAs back-to-back
#pragma unroll
    for (int nt = 0; nt < 8; ++nt)
      acc[nt] = __builtin_amdgcn_wmma_f32_16x16x32_bf16(
          false, af, false, bfr[nt], (short)0, acc[nt], false, false);
  }

  // Dump accumulators to LDS: VGPR g, lane -> row = g + kh*8, col = n-tile col
  const int n0 = wave * 128;
#pragma unroll
  for (int nt = 0; nt < 8; ++nt)
#pragma unroll
    for (int g = 0; g < 8; ++g)
      tile[(g + kh * 8) * ldt + n0 + nt * 16 + row_in] = acc[nt][g];
  __syncthreads();

  // ---- row epilogue: rpt threads cooperate on one output row ----
  const int rpt = blockDim.x >> 4;   // 16 rows per workgroup
  const int row = tid / rpt;
  const int t = tid % rpt;
  const long long gm = m0 + row;

  float s1v = 0.f, s2v = 0.f;
  if (ln_g) {
    for (int c = t; c < ncols; c += rpt) {
      float x = tile[row * ldt + c] + (bias ? bias[c] : 0.f);
      s1v += x; s2v += x * x;
    }
    for (int o = 1; o < rpt; o <<= 1) {
      s1v += __shfl_xor(s1v, o, 32);
      s2v += __shfl_xor(s2v, o, 32);
    }
  }
  float mu = s1v / (float)ncols;
  float var = s2v / (float)ncols - mu * mu;
  float rstd = rsqrtf(var + 1e-5f);
  float rs = row_scale ? row_scale[gm] : 1.f;

  float dacc = 0.f;
  for (int c = t; c < N; c += rpt) {
    float y = 0.f;
    if (c < ncols) {
      float x = tile[row * ldt + c] + (bias ? bias[c] : 0.f);
      if (ln_g) x = (x - mu) * rstd * ln_g[c] + ln_b[c];
      if (act == 1) x = fmaxf(x, 0.f);
      else if (act == 2) x = tanhf(x);
      x *= rs;
      y = x;
      if (out_f32) out_f32[gm * ldof + c] = x;
      if (atom_out) atomicAdd(&atom_out[(gm / atom_div) * ld_atom + c], x);
      if (dotw) dacc += x * dotw[c];
    }
    if (out_bf) out_bf[aswz(gm, c, (int)(ldob >> 5))] = f2bf(y);
  }
  if (dot_out) {
    for (int o = 1; o < rpt; o <<= 1) dacc += __shfl_xor(dacc, o, 32);
    if (t == 0) dot_out[gm] = sigmoidf_(dacc + (dotb ? dotb[0] : 0.f));
  }
}

// ---------------------------------------------------------------------------
extern "C" void kernel_launch(void* const* d_in, const int* in_sizes, int n_in,
                              void* d_out, int out_size, void* d_ws, size_t ws_size,
                              hipStream_t stream) {
  (void)in_sizes; (void)n_in; (void)out_size; (void)ws_size;
  const float* state    = (const float*)d_in[0];
  const float* actions  = (const float*)d_in[1];
  const float* ac_fc1_w = (const float*)d_in[2];
  const float* ac_fc1_b = (const float*)d_in[3];
  const float* ac_ln1_g = (const float*)d_in[4];
  const float* ac_ln1_b = (const float*)d_in[5];
  const float* ac_fc2_w = (const float*)d_in[6];
  const float* ac_fc2_b = (const float*)d_in[7];
  const float* ac_lnf2_w= (const float*)d_in[8];
  const float* ac_lnf2_b= (const float*)d_in[9];
  const float* ac_fc3_w = (const float*)d_in[10];
  const float* ac_fc3_b = (const float*)d_in[11];
  const float* fc1_w = (const float*)d_in[12];
  const float* fc1_b = (const float*)d_in[13];
  const float* ln1_g = (const float*)d_in[14];
  const float* ln1_b = (const float*)d_in[15];
  const float* fc2_w = (const float*)d_in[16];
  const float* fc2_b = (const float*)d_in[17];
  const float* ln2_g = (const float*)d_in[18];
  const float* ln2_b = (const float*)d_in[19];
  const float* fc3_w = (const float*)d_in[20];
  const float* fc3_b = (const float*)d_in[21];
  const float* ln3_g = (const float*)d_in[22];
  const float* ln3_b = (const float*)d_in[23];
  const float* fc4_w = (const float*)d_in[24];
  const float* fc4_b = (const float*)d_in[25];
  const float* ln4_g = (const float*)d_in[26];
  const float* ln4_b = (const float*)d_in[27];
  const float* fc5_w = (const float*)d_in[28];
  const float* fc5_b = (const float*)d_in[29];
  const float* fc6_w = (const float*)d_in[30];
  const float* fc6_b = (const float*)d_in[31];
  const float* ln6_g = (const float*)d_in[32];
  const float* ln6_b = (const float*)d_in[33];

  const int Mbk   = 8192;     // B*K rows
  const int Mcore = 122880;   // B*K*(K-1) rows

  char* ws = (char*)d_ws;
  size_t off = 0;
  auto alloc = [&](size_t bytes) -> char* {
    char* p = ws + off;
    off = (off + bytes + 255) & ~(size_t)255;
    return p;
  };
  u16*   act_bf  = (u16*)alloc((size_t)512 * 64 * 2);
  u16*   st_bf   = (u16*)alloc((size_t)Mbk * HP * 2);
  u16*   Wac1    = (u16*)alloc((size_t)HP * 64 * 2);
  u16*   Wac2    = (u16*)alloc((size_t)HP * 1024 * 2);
  u16*   Wlnf2   = (u16*)alloc((size_t)HP * HP * 2);
  u16*   Wfc1c   = (u16*)alloc((size_t)HP * HP * 2);
  u16*   Wfc2t   = (u16*)alloc((size_t)HP * HP * 2);
  u16*   Wfc2b   = (u16*)alloc((size_t)HP * HP * 2);
  u16*   Wfc3c   = (u16*)alloc((size_t)HP * HP * 2);
  u16*   Wfc4c   = (u16*)alloc((size_t)128 * HP * 2);
  u16*   Wfc6c   = (u16*)alloc((size_t)HP * 1024 * 2);
  u16*   a_bf    = (u16*)alloc((size_t)512 * HP * 2);
  float* a_f32   = (float*)alloc((size_t)512 * HP * 4);
  float* attA    = (float*)alloc((size_t)Mbk * 4);
  u16*   T_bf    = (u16*)alloc((size_t)Mbk * HP * 2);
  u16*   st2_bf  = (u16*)alloc((size_t)Mbk * HP * 2);
  u16*   s1_bf   = (u16*)alloc((size_t)Mbk * HP * 2);
  float* Pb      = (float*)alloc((size_t)Mbk * HP * 4);
  float* Qb      = (float*)alloc((size_t)Mbk * HP * 4);
  u16*   core_bf = (u16*)alloc((size_t)Mcore * HP * 2);
  float* attS    = (float*)alloc((size_t)Mcore * 4);
  float* esum    = (float*)alloc((size_t)Mbk * HP * 4);
  u16*   es_bf   = (u16*)alloc((size_t)Mbk * HP * 2);

  auto gemm = [&](const u16* A1, int nkt1, const u16* A2, int rdiv,
                  const u16* B, int Ktot,
                  int M, int N, int ncols, const float* bias,
                  const float* g, const float* bv, int act, const float* rsc,
                  u16* obf, long long ldob, float* of32, long long ldof,
                  float* aout, int adiv, long long lda_atom,
                  const float* dw, const float* db, float* dout) {
    dim3 grid(M / 16), block((N / 128) * 32);
    size_t sh = (size_t)16 * (N + 4) * sizeof(float);
    gemm_epi_kernel<<<grid, block, sh, stream>>>(
        A1, nkt1, A2, rdiv, B, Ktot, N, ncols, bias, g, bv, act,
        rsc, obf, ldob, of32, ldof, aout, adiv, lda_atom, dw, db, dout);
  };
  auto cdiv = [](long long a, long long b) { return (unsigned)((a + b - 1) / b); };

  // ---- pack activations into A-fragment-major bf16 ----
  conv_act_kernel<<<cdiv(512 * 64, 256), 256, 0, stream>>>(actions, 64, 64, act_bf, 64, 512);
  conv_act_kernel<<<cdiv((long long)Mbk * HP, 256), 256, 0, stream>>>(state, HH, HH, st_bf, HP, Mbk);
  // ---- pack weights into B-fragment-major bf16 ----
  // args: src, src_ld, kvalid, nvalid, dst, nkt_total, kt_off, nkt_zone, ntn
  conv_wfrag_kernel<<<cdiv(HP * 64, 256), 256, 0, stream>>>(ac_fc1_w, HH, 64, HH, Wac1, 2, 0, 2, 32);
  conv_wfrag_kernel<<<cdiv(HP * HP, 256), 256, 0, stream>>>(ac_fc2_w, HH, HH, HH, Wac2, 32, 0, 16, 32);
  conv_wfrag_kernel<<<cdiv(HP * HP, 256), 256, 0, stream>>>(ac_fc2_w + (size_t)HH * HH, HH, HH, HH, Wac2, 32, 16, 16, 32);
  conv_wfrag_kernel<<<cdiv(HP * HP, 256), 256, 0, stream>>>(ac_lnf2_w, HH, HH, HH, Wlnf2, 16, 0, 16, 32);
  conv_wfrag_kernel<<<cdiv(HP * HP, 256), 256, 0, stream>>>(fc1_w, HH, HH, HH, Wfc1c, 16, 0, 16, 32);
  conv_wfrag_kernel<<<cdiv(HP * HP, 256), 256, 0, stream>>>(fc2_w, HH, HH, HH, Wfc2t, 16, 0, 16, 32);
  conv_wfrag_kernel<<<cdiv(HP * HP, 256), 256, 0, stream>>>(fc2_w + (size_t)HH * HH, HH, HH, HH, Wfc2b, 16, 0, 16, 32);
  conv_wfrag_kernel<<<cdiv(HP * HP, 256), 256, 0, stream>>>(fc3_w, HH, HH, HH, Wfc3c, 16, 0, 16, 32);
  conv_wfrag_kernel<<<cdiv(128 * HP, 256), 256, 0, stream>>>(fc4_w, 100, HH, 100, Wfc4c, 16, 0, 16, 8);
  conv_wfrag_kernel<<<cdiv(HP * HP, 256), 256, 0, stream>>>(fc6_w, HH, HH, HH, Wfc6c, 32, 0, 16, 32);
  conv_wfrag_kernel<<<cdiv(HP * HP, 256), 256, 0, stream>>>(fc6_w + (size_t)HH * HH, HH, HH, HH, Wfc6c, 32, 16, 16, 32);

  // G1: a = LN(actions @ ac_fc1_w + b)  -> a_bf + a_f32
  gemm(act_bf, 2, nullptr, 1, Wac1, 64, 512, HP, HH,
       ac_fc1_b, ac_ln1_g, ac_ln1_b, 0, nullptr,
       a_bf, HP, a_f32, HP, nullptr, 1, 0, nullptr, nullptr, nullptr);

  // attention_a = sigmoid(concat_a @ ac_fc3_w + b3)
  att_kernel<<<Mbk / 8, 256, 0, stream>>>(state, a_f32, ac_fc3_w, ac_fc3_b, attA, Mbk);

  // G2: T = concat_a @ ac_fc2_w + b2   (dual-A concat GEMM; a broadcast via rowdiv=16)
  gemm(st_bf, 16, a_bf, 16, Wac2, 1024, Mbk, HP, HH,
       ac_fc2_b, nullptr, nullptr, 0, nullptr,
       T_bf, HP, nullptr, 0, nullptr, 1, 0, nullptr, nullptr, nullptr);

  // G3: state2 = (T @ ac_lnf2_w + b) * attention_a
  gemm(T_bf, 16, nullptr, 1, Wlnf2, 512, Mbk, HP, HH,
       ac_lnf2_b, nullptr, nullptr, 0, attA,
       st2_bf, HP, nullptr, 0, nullptr, 1, 0, nullptr, nullptr, nullptr);

  // G4: s1 = relu(LN(state2 @ fc1_w + b))
  gemm(st2_bf, 16, nullptr, 1, Wfc1c, 512, Mbk, HP, HH,
       fc1_b, ln1_g, ln1_b, 1, nullptr,
       s1_bf, HP, nullptr, 0, nullptr, 1, 0, nullptr, nullptr, nullptr);

  // G5/G6: P = s1 @ fc2_w[:500], Q = s1 @ fc2_w[500:]   (pair-GEMM factorization)
  gemm(s1_bf, 16, nullptr, 1, Wfc2t, 512, Mbk, HP, HH,
       nullptr, nullptr, nullptr, 0, nullptr,
       nullptr, HP, Pb, HP, nullptr, 1, 0, nullptr, nullptr, nullptr);
  gemm(s1_bf, 16, nullptr, 1, Wfc2b, 512, Mbk, HP, HH,
       nullptr, nullptr, nullptr, 0, nullptr,
       nullptr, HP, Qb, HP, nullptr, 1, 0, nullptr, nullptr, nullptr);

  // core = relu(LN(P[bi] + Q[bjj] + fc2_b))
  core_kernel<<<Mcore / 8, 256, 0, stream>>>(Pb, Qb, fc2_b, ln2_g, ln2_b, core_bf, Mcore);

  // G7: att_small = sigmoid(tanh(LN(core @ fc4_w + b)) @ fc5_w + b5)  (fused dot epilogue)
  gemm(core_bf, 16, nullptr, 1, Wfc4c, 512, Mcore, 128, 100,
       fc4_b, ln4_g, ln4_b, 2, nullptr,
       nullptr, 128, nullptr, 0, nullptr, 1, 0, fc5_w, fc5_b, attS);

  // effect_sum = 0
  zero_kernel<<<cdiv((long long)Mbk * HP, 256), 256, 0, stream>>>(esum, (long long)Mbk * HP);

  // G8: context = relu(LN(core @ fc3_w + b)); effect_sum[r/15] += context * att_small[r]
  gemm(core_bf, 16, nullptr, 1, Wfc3c, 512, Mcore, HP, HH,
       fc3_b, ln3_g, ln3_b, 1, attS,
       nullptr, HP, nullptr, 0, esum, 15, HP, nullptr, nullptr, nullptr);

  // effect_sum -> bf16 (A-fragment-major)
  conv_act_kernel<<<cdiv((long long)Mbk * HP, 256), 256, 0, stream>>>(esum, HP, HP, es_bf, HP, Mbk);

  // G9: h_new = LN([s1 | effect_sum] @ fc6_w + b)  -> d_out (8192 x 500, fp32)
  gemm(s1_bf, 16, es_bf, 1, Wfc6c, 1024, Mbk, HP, HH,
       fc6_b, ln6_g, ln6_b, 0, nullptr,
       nullptr, HP, (float*)d_out, HH, nullptr, 1, 0, nullptr, nullptr, nullptr);
}
